// TokenSeqLayer_11819749998990
// MI455X (gfx1250) — matrix operand order repaired and checked by reading, not compiled, not run on hardware
//
#include <hip/hip_runtime.h>
#include <hip/hip_bf16.h>
#include <math.h>

#define B_ 16
#define S_ 2048
#define D_ 256
#define LN_EPS 1e-5f
#define NEG_BIG -1.0e7f

typedef __attribute__((ext_vector_type(16))) __bf16 v16bf;
typedef __attribute__((ext_vector_type(8)))  float  v8f;
typedef __attribute__((ext_vector_type(4)))  int    v4i;

// ---------------- async global->LDS staging (CDNA5), with portable fallback ----------
#if defined(__has_builtin)
#if __has_builtin(__builtin_amdgcn_global_load_async_to_lds_b128) && \
    __has_builtin(__builtin_amdgcn_s_wait_asynccnt)
#define USE_ASYNC_LDS 1
#endif
#endif

__device__ __forceinline__ void copy16_g2l(const __bf16* g, __bf16* l) {
#if defined(USE_ASYNC_LDS)
  __builtin_amdgcn_global_load_async_to_lds_b128(
      (__attribute__((address_space(1))) v4i*)g,
      (__attribute__((address_space(3))) v4i*)l, 0, 0);
#else
  *reinterpret_cast<uint4*>(l) = *reinterpret_cast<const uint4*>(g);
#endif
}
__device__ __forceinline__ void wait_stage() {
#if defined(USE_ASYNC_LDS)
  __builtin_amdgcn_s_wait_asynccnt(0);
#endif
}

// stage 32 rows (16KB, contiguous) with 128 threads: 8 x 16B per thread
__device__ __forceinline__ void stage_tile(const __bf16* __restrict__ gsrc,
                                           __bf16* __restrict__ ldst, int tid) {
#pragma unroll
  for (int i = 0; i < 8; ++i) {
    const int ofs = (tid + i * 128) * 8;   // bf16 elements, 16B chunks
    copy16_g2l(gsrc + ofs, ldst + ofs);
  }
}

// ---------------- WMMA fragment loaders (CDNA5 ISA 7.12.2 layouts) ----------------
// A 16x32 bf16: lane {m=lane&15, half=lane>>4}; elems 0..7 = K[half*8..+7],
// elems 8..15 = K[16+half*8..+7] -> two 16B loads from a row-major row.
__device__ __forceinline__ v16bf load_fragA(const __bf16* row, int chunk, int half) {
  v16bf a;
  uint4* d = reinterpret_cast<uint4*>(&a);
  d[0] = *reinterpret_cast<const uint4*>(row + chunk * 32 + half * 8);
  d[1] = *reinterpret_cast<const uint4*>(row + chunk * 32 + 16 + half * 8);
  return a;
}
// B 32x16 bf16 (col n = h row): lane {n=lane&15, half}; elems 0..7 = K[half*8..+7],
// elems 8..15 = K[16+half*8..+7].
__device__ __forceinline__ v16bf load_fragB(const __bf16* row, int chunk, int half) {
  v16bf b;
  uint4* d = reinterpret_cast<uint4*>(&b);
  d[0] = *reinterpret_cast<const uint4*>(row + chunk * 32 + half * 8);
  d[1] = *reinterpret_cast<const uint4*>(row + chunk * 32 + 16 + half * 8);
  return b;
}

// ---------------- Kernel 1: LayerNorm -> h (f32 + bf16), mask bias array -------------
__global__ void ln_kernel(const float* __restrict__ x, const int* __restrict__ mask,
                          const float* __restrict__ gamma, const float* __restrict__ beta,
                          float* __restrict__ hf, __bf16* __restrict__ hb,
                          float* __restrict__ neg) {
  const int row = blockIdx.x;          // b*S + s
  const int d = threadIdx.x;           // 0..255
  const float v = x[(size_t)row * D_ + d];
  __shared__ float red1[8], red2[8];
  const int lane = threadIdx.x & 31, w = threadIdx.x >> 5;
  float s1 = v, s2 = v * v;
  for (int off = 16; off; off >>= 1) {
    s1 += __shfl_down(s1, off, 32);
    s2 += __shfl_down(s2, off, 32);
  }
  if (lane == 0) { red1[w] = s1; red2[w] = s2; }
  __syncthreads();
  if (w == 0) {
    float a = (lane < 8) ? red1[lane] : 0.0f;
    float bq = (lane < 8) ? red2[lane] : 0.0f;
    for (int off = 4; off; off >>= 1) {
      a += __shfl_down(a, off, 32);
      bq += __shfl_down(bq, off, 32);
    }
    if (lane == 0) { red1[0] = a; red2[0] = bq; }
  }
  __syncthreads();
  const float mean = red1[0] * (1.0f / D_);
  const float var = red2[0] * (1.0f / D_) - mean * mean;
  const float hn = (v - mean) * rsqrtf(var + LN_EPS) * gamma[d] + beta[d];
  hf[(size_t)row * D_ + d] = hn;
  hb[(size_t)row * D_ + d] = (__bf16)hn;
  if (threadIdx.x == 0) neg[row] = mask[row] ? 0.0f : NEG_BIG;
}

// online softmax update for one 16x16 sub-tile / column-set
__device__ __forceinline__ void stats_update(const v8f& acc, const float* nq, float negk,
                                             float& mrun, float& zrun) {
  float sv[8];
  float tmax = -3.0e38f;
#pragma unroll
  for (int r = 0; r < 8; ++r) {
    sv[r] = acc[r] * 0.25f + nq[r] + negk;   // scale 1/d^0.25 = 1/4, additive mask bias
    tmax = fmaxf(tmax, sv[r]);
  }
  tmax = fmaxf(tmax, __shfl_xor(tmax, 16, 32));
  const float newm = fmaxf(mrun, tmax);
  float ls = 0.0f;
#pragma unroll
  for (int r = 0; r < 8; ++r) ls += __expf(sv[r] - newm);
  ls += __shfl_xor(ls, 16, 32);
  zrun = zrun * __expf(mrun - newm) + ls;
  mrun = newm;
}

// ---------------- Kernel 2: per-column online softmax stats (m_k, 1/Z_k) -------------
// Block = 128 threads = 4 waves; WG owns 128 k-columns (32/wave, two 16-col sets with
// stationary B in VGPRs); streams q in 32-row tiles double-buffered through LDS.
__global__ void __launch_bounds__(128)
colstats_kernel(const __bf16* __restrict__ hb, const float* __restrict__ neg,
                float* __restrict__ Mcol, float* __restrict__ Zinv) {
  __shared__ __bf16 tile[2][32 * D_];            // 2 x 16KB
  const int b = blockIdx.y;
  const int wave = threadIdx.x >> 5;
  const int lane = threadIdx.x & 31;
  const int half = lane >> 4;
  const int n = lane & 15;
  const int k0 = blockIdx.x * 128 + wave * 32;
  const size_t base = (size_t)b * S_;

  // stationary operand: two 16-column B sets, full K=256, in VGPRs
  v16bf Bf[2][8];
  float negk[2];
#pragma unroll
  for (int cs = 0; cs < 2; ++cs) {
    const int kcol = k0 + cs * 16 + n;
    const __bf16* krow = hb + (base + kcol) * (size_t)D_;
#pragma unroll
    for (int c = 0; c < 8; ++c) Bf[cs][c] = load_fragB(krow, c, half);
    negk[cs] = neg[base + kcol];
  }

  float mrun[2] = {-3.0e38f, -3.0e38f};
  float zrun[2] = {0.0f, 0.0f};

  stage_tile(hb + base * (size_t)D_, &tile[0][0], threadIdx.x);   // q-tile 0
  for (int t = 0; t < S_ / 32; ++t) {
    wait_stage();
    __syncthreads();                              // staged tile visible to all waves
    if (t + 1 < S_ / 32)
      stage_tile(hb + (base + (size_t)(t + 1) * 32) * (size_t)D_,
                 &tile[(t + 1) & 1][0], threadIdx.x);
    const __bf16* lt = &tile[t & 1][0];
    const int q0 = t * 32;
#pragma unroll
    for (int sub = 0; sub < 2; ++sub) {
      const __bf16* arow = lt + (sub * 16 + n) * D_;
      v8f acc0 = {}, acc1 = {};
#pragma unroll
      for (int c = 0; c < 8; ++c) {
        v16bf Af = load_fragA(arow, c, half);
        acc0 = __builtin_amdgcn_wmma_f32_16x16x32_bf16(false, Af, false, Bf[0][c],
                                                       (short)0, acc0, false, false);
        acc1 = __builtin_amdgcn_wmma_f32_16x16x32_bf16(false, Af, false, Bf[1][c],
                                                       (short)0, acc1, false, false);
      }
      float nq[8];
#pragma unroll
      for (int r = 0; r < 8; ++r) nq[r] = neg[base + q0 + sub * 16 + half * 8 + r];
      stats_update(acc0, nq, negk[0], mrun[0], zrun[0]);
      stats_update(acc1, nq, negk[1], mrun[1], zrun[1]);
    }
  }
  if (half == 0) {
#pragma unroll
    for (int cs = 0; cs < 2; ++cs) {
      const int kcol = k0 + cs * 16 + n;
      Mcol[base + kcol] = mrun[cs];
      Zinv[base + kcol] = (negk[cs] == 0.0f && zrun[cs] > 0.0f) ? (1.0f / zrun[cs]) : 0.0f;
    }
  }
}

// ---------------- Kernel 3: w[q] = sum_k exp(s - m_k) * zinv_k -----------------------
// Block = 128 threads = 4 waves; WG owns 128 q-rows (32/wave, stationary A in VGPRs);
// streams k in 32-row tiles double-buffered through LDS.
__global__ void __launch_bounds__(128)
wq_kernel(const __bf16* __restrict__ hb, const float* __restrict__ neg,
          const float* __restrict__ Mcol, const float* __restrict__ Zinv,
          float* __restrict__ W) {
  __shared__ __bf16 tile[2][32 * D_];
  const int b = blockIdx.y;
  const int wave = threadIdx.x >> 5;
  const int lane = threadIdx.x & 31;
  const int half = lane >> 4;
  const int n = lane & 15;
  const int q0 = blockIdx.x * 128 + wave * 32;
  const size_t base = (size_t)b * S_;

  // stationary operand: two 16-row A sets, full K=256, in VGPRs
  v16bf Af[2][8];
  float nqv[2][8];
#pragma unroll
  for (int cs = 0; cs < 2; ++cs) {
    const __bf16* arow = hb + (base + q0 + cs * 16 + n) * (size_t)D_;
#pragma unroll
    for (int c = 0; c < 8; ++c) Af[cs][c] = load_fragA(arow, c, half);
#pragma unroll
    for (int r = 0; r < 8; ++r) nqv[cs][r] = neg[base + q0 + cs * 16 + half * 8 + r];
  }

  float wacc[2][8] = {{0}};

  stage_tile(hb + base * (size_t)D_, &tile[0][0], threadIdx.x);   // k-tile 0
  for (int t = 0; t < S_ / 32; ++t) {
    wait_stage();
    __syncthreads();
    if (t + 1 < S_ / 32)
      stage_tile(hb + (base + (size_t)(t + 1) * 32) * (size_t)D_,
                 &tile[(t + 1) & 1][0], threadIdx.x);
    const __bf16* lt = &tile[t & 1][0];
    const int kt = t * 32;
#pragma unroll
    for (int sub = 0; sub < 2; ++sub) {
      const __bf16* krow = lt + (sub * 16 + n) * D_;
      v8f acc0 = {}, acc1 = {};
#pragma unroll
      for (int c = 0; c < 8; ++c) {
        v16bf Bf = load_fragB(krow, c, half);
        acc0 = __builtin_amdgcn_wmma_f32_16x16x32_bf16(false, Af[0][c], false, Bf,
                                                       (short)0, acc0, false, false);
        acc1 = __builtin_amdgcn_wmma_f32_16x16x32_bf16(false, Af[1][c], false, Bf,
                                                       (short)0, acc1, false, false);
      }
      const int kcol = kt + sub * 16 + n;
      const float mcol = Mcol[base + kcol];
      const float zinv = Zinv[base + kcol];
      const float nk = neg[base + kcol];
#pragma unroll
      for (int r = 0; r < 8; ++r) {
        // clamp at 0: valid entries have s-m_k <= 0; keeps masked rows/cols finite
        // (their exp underflows to exactly 0 or is killed by zinv==0).
        wacc[0][r] += zinv * __expf(fminf(acc0[r] * 0.25f + nqv[0][r] + nk - mcol, 0.0f));
        wacc[1][r] += zinv * __expf(fminf(acc1[r] * 0.25f + nqv[1][r] + nk - mcol, 0.0f));
      }
    }
  }
  // sum over k columns: reduce across the 16 lanes of each half
#pragma unroll
  for (int cs = 0; cs < 2; ++cs) {
#pragma unroll
    for (int r = 0; r < 8; ++r) {
      float v = wacc[cs][r];
      v += __shfl_xor(v, 1, 32);
      v += __shfl_xor(v, 2, 32);
      v += __shfl_xor(v, 4, 32);
      v += __shfl_xor(v, 8, 32);
      if (n == 0) W[base + q0 + cs * 16 + half * 8 + r] = v;
    }
  }
}

// ---------------- Kernel 4: out = normalize( sum_s mask*x + w*h ) --------------------
__global__ void finalize_kernel(const float* __restrict__ x, const int* __restrict__ mask,
                                const float* __restrict__ hf, const float* __restrict__ W,
                                float* __restrict__ out) {
  const int b = blockIdx.x;
  const int d = threadIdx.x;          // 0..255
  const size_t base = (size_t)b * S_;
  float acc = 0.0f;
  for (int s = 0; s < S_; ++s) {
    const float xv = x[(base + s) * (size_t)D_ + d];
    const float hv = hf[(base + s) * (size_t)D_ + d];
    const float wv = W[base + s];
    const float mm = mask[base + s] ? 1.0f : 0.0f;
    acc += mm * xv + wv * hv;
  }
  __shared__ float red[8];
  const int lane = threadIdx.x & 31, w = threadIdx.x >> 5;
  float sq = acc * acc;
  for (int off = 16; off; off >>= 1) sq += __shfl_down(sq, off, 32);
  if (lane == 0) red[w] = sq;
  __syncthreads();
  if (w == 0) {
    float t = (lane < 8) ? red[lane] : 0.0f;
    for (int off = 4; off; off >>= 1) t += __shfl_down(t, off, 32);
    if (lane == 0) red[0] = t;
  }
  __syncthreads();
  out[(size_t)b * D_ + d] = acc * rsqrtf(fmaxf(red[0], 1e-30f));
}

extern "C" void kernel_launch(void* const* d_in, const int* in_sizes, int n_in,
                              void* d_out, int out_size, void* d_ws, size_t ws_size,
                              hipStream_t stream) {
  const float* x     = (const float*)d_in[0];
  const int*   mask  = (const int*)d_in[1];
  const float* gamma = (const float*)d_in[2];
  const float* beta  = (const float*)d_in[3];
  float* out = (float*)d_out;

  char* ws = (char*)d_ws;
  size_t off = 0;
  float*  Hf   = (float*)(ws + off);  off += (size_t)B_ * S_ * D_ * sizeof(float);
  __bf16* Hb   = (__bf16*)(ws + off); off += (size_t)B_ * S_ * D_ * sizeof(__bf16);
  float*  Mcol = (float*)(ws + off);  off += (size_t)B_ * S_ * sizeof(float);
  float*  Zinv = (float*)(ws + off);  off += (size_t)B_ * S_ * sizeof(float);
  float*  W    = (float*)(ws + off);  off += (size_t)B_ * S_ * sizeof(float);
  float*  Neg  = (float*)(ws + off);  off += (size_t)B_ * S_ * sizeof(float);
  (void)ws_size; (void)in_sizes; (void)n_in; (void)out_size;

  ln_kernel<<<dim3(B_ * S_), dim3(D_), 0, stream>>>(x, mask, gamma, beta, Hf, Hb, Neg);
  colstats_kernel<<<dim3(S_ / 128, B_), dim3(128), 0, stream>>>(Hb, Neg, Mcol, Zinv);
  wq_kernel<<<dim3(S_ / 128, B_), dim3(128), 0, stream>>>(Hb, Neg, Mcol, Zinv, W);
  finalize_kernel<<<dim3(B_), dim3(D_), 0, stream>>>(x, mask, Hf, W, out);
}
